// ResidualAttentionBlock_expert_9689446220386
// MI455X (gfx1250) — matrix operand order repaired and checked by reading, not compiled, not run on hardware
//
#include <hip/hip_runtime.h>

typedef __bf16 v16bf __attribute__((ext_vector_type(16)));
typedef __bf16 v8bf  __attribute__((ext_vector_type(8)));
typedef float  v8f   __attribute__((ext_vector_type(8)));

#define Dm   768
#define Hh   12
#define HDm  64
#define Mm   3072
#define Ee   36
#define Ll   293
#define Nn   64
#define Ttok (Ll * Nn)   /* 18752, multiple of 16 */

static __device__ __forceinline__ __bf16 f2bf(float f) {
  unsigned u = __builtin_bit_cast(unsigned, f);
  u += 0x7FFFu + ((u >> 16) & 1u);             // round-to-nearest-even
  unsigned short h = (unsigned short)(u >> 16);
  return __builtin_bit_cast(__bf16, h);
}

static __device__ __forceinline__ float gelu_exact(float x) {
  return 0.5f * x * (1.0f + erff(x * 0.70710678118654752f));
}

// ---------------------------------------------------------------------------
// Elementwise fp32 -> bf16 (weights pre-pass). 8 elements per thread.
// ---------------------------------------------------------------------------
__global__ __launch_bounds__(256) void cvtbf_kernel(
    const float* __restrict__ in, __bf16* __restrict__ out, long long n)
{
  long long i = ((long long)blockIdx.x * 256 + threadIdx.x) * 8;
  if (i >= n) return;
  if (i + 8 <= n) {
    float4 a = *(const float4*)(in + i);
    float4 b = *(const float4*)(in + i + 4);
    v8bf o;
    o[0] = f2bf(a.x); o[1] = f2bf(a.y); o[2] = f2bf(a.z); o[3] = f2bf(a.w);
    o[4] = f2bf(b.x); o[5] = f2bf(b.y); o[6] = f2bf(b.z); o[7] = f2bf(b.w);
    *(v8bf*)(out + i) = o;
  } else {
    for (long long j = i; j < n; j++) out[j] = f2bf(in[j]);
  }
}

// ---------------------------------------------------------------------------
// LayerNorm: one wave per token; lane owns 24 contiguous columns.
// fp32 math, bf16 output (consumed only by WMMA GEMMs).
// ---------------------------------------------------------------------------
__global__ __launch_bounds__(256) void ln_kernel(
    const float* __restrict__ x, const float* __restrict__ w,
    const float* __restrict__ b, __bf16* __restrict__ out, int ntok)
{
  int wid  = threadIdx.x >> 5;
  int lane = threadIdx.x & 31;
  int tok  = blockIdx.x * (blockDim.x >> 5) + wid;
  if (tok >= ntok) return;
  const int CPL = Dm / 32;                     // 24 columns per lane
  const float* xr = x + (size_t)tok * Dm + lane * CPL;
  float vals[CPL];
  float s = 0.f;
#pragma unroll
  for (int i = 0; i < CPL; i++) { vals[i] = xr[i]; s += vals[i]; }
#pragma unroll
  for (int off = 16; off > 0; off >>= 1) s += __shfl_xor(s, off, 32);
  float mean = s * (1.0f / Dm);
  float var = 0.f;
#pragma unroll
  for (int i = 0; i < CPL; i++) { float d = vals[i] - mean; var += d * d; }
#pragma unroll
  for (int off = 16; off > 0; off >>= 1) var += __shfl_xor(var, off, 32);
  float inv = rsqrtf(var * (1.0f / Dm) + 1e-5f);
  const float* wl = w + lane * CPL;
  const float* bl = b + lane * CPL;
  __bf16* orow = out + (size_t)tok * Dm + lane * CPL;
  v8bf o[CPL / 8];
#pragma unroll
  for (int i = 0; i < CPL; i++)
    o[i >> 3][i & 7] = f2bf((vals[i] - mean) * inv * wl[i] + bl[i]);
#pragma unroll
  for (int i = 0; i < CPL / 8; i++) *(v8bf*)(orow + i * 8) = o[i];
}

// ---------------------------------------------------------------------------
// bf16-WMMA GEMM:  out = act(A @ W^T + bias) [+ res]
//   A: rows x K bf16, W: Dout x K bf16, bias/res fp32.
//   Cf (fp32) and/or Cb (bf16) outputs, ldc = Dout.
//   rows % 16 == 0 (exact grid), Dout % 64 == 0, K % 32 == 0.
//   Batched via gridDim.z with element strides.
// One wave per block; wave computes a 16x64 tile; inner loop = 10 vector
// loads + 4 v_wmma (no per-element conversion).
// ---------------------------------------------------------------------------
__global__ __launch_bounds__(32) void gemm_bf16_kernel(
    const __bf16* __restrict__ A, const __bf16* __restrict__ W,
    const float* __restrict__ bias, const float* __restrict__ res,
    float* __restrict__ Cf, __bf16* __restrict__ Cb, int K, int ldc,
    long long abs_, long long wbs, long long bbs, long long rbs, long long cbs,
    int act)
{
  long long z = blockIdx.z;
  A    += z * abs_;
  W    += z * wbs;
  bias += z * bbs;
  if (res) res += z * rbs;
  if (Cf)  Cf  += z * cbs;
  if (Cb)  Cb  += z * cbs;

  int lane = threadIdx.x & 31;
  int hi   = (lane >> 4) & 1;
  int m    = lane & 15;
  int row  = blockIdx.x * 16 + m;
  int col0 = blockIdx.y * 64;

  const __bf16* arow = A + (size_t)row * K;
  const __bf16* wrow = W + (size_t)(col0 + m) * K;
  int aoff = hi ? 16 : 0;   // A frag: lo lanes K 0..15 contiguous, hi 16..31
  int boff = hi ? 16 : 0;   // B frag: lo lanes K 0..15 contiguous, hi 16..31

  v8f acc0 = {}, acc1 = {}, acc2 = {}, acc3 = {};
  for (int kk = 0; kk < K; kk += 32) {
    v8bf alo = *(const v8bf*)(arow + kk + aoff);
    v8bf ahi = *(const v8bf*)(arow + kk + 8 + aoff);
    v16bf a = __builtin_shufflevector(alo, ahi, 0,1,2,3,4,5,6,7,8,9,10,11,12,13,14,15);
    const __bf16* wp = wrow + kk + boff;
    v16bf b0 = *(const v16bf*)(wp);
    v16bf b1 = *(const v16bf*)(wp + (size_t)16 * K);
    v16bf b2 = *(const v16bf*)(wp + (size_t)32 * K);
    v16bf b3 = *(const v16bf*)(wp + (size_t)48 * K);
    acc0 = __builtin_amdgcn_wmma_f32_16x16x32_bf16(false, a, false, b0, (short)0, acc0, false, false);
    acc1 = __builtin_amdgcn_wmma_f32_16x16x32_bf16(false, a, false, b1, (short)0, acc1, false, false);
    acc2 = __builtin_amdgcn_wmma_f32_16x16x32_bf16(false, a, false, b2, (short)0, acc2, false, false);
    acc3 = __builtin_amdgcn_wmma_f32_16x16x32_bf16(false, a, false, b3, (short)0, acc3, false, false);
  }

  int rbase = blockIdx.x * 16 + hi * 8;   // C/D layout: M = vgpr + 8*hi, N = lane&15
#pragma unroll
  for (int vg = 0; vg < 8; vg++) {
    size_t ro = (size_t)(rbase + vg) * ldc;
#pragma unroll
    for (int c = 0; c < 4; c++) {
      int col = col0 + c * 16 + m;
      float val = (c == 0 ? acc0[vg] : c == 1 ? acc1[vg] : c == 2 ? acc2[vg] : acc3[vg]);
      val += bias[col];
      if (act == 1) val = gelu_exact(val);
      if (res) val += res[ro + col];
      if (Cf) Cf[ro + col] = val;
      if (Cb) Cb[ro + col] = f2bf(val);
    }
  }
}

// ---------------------------------------------------------------------------
// Attention: one wave per (query-tile of 16, head, batch); q/k/v/ao are bf16.
// Full-row softmax staged in LDS (16 rows x 320 padded key columns).
// ---------------------------------------------------------------------------
__global__ __launch_bounds__(32) void attn_kernel(
    const __bf16* __restrict__ q, const __bf16* __restrict__ k,
    const __bf16* __restrict__ v, __bf16* __restrict__ ao)
{
  __shared__ float S[16 * 320];
  int lane = threadIdx.x;
  int m  = lane & 15;
  int hi = (lane >> 4) & 1;
  int qt = blockIdx.x, h = blockIdx.y, n = blockIdx.z;
  int aoff = hi ? 16 : 0;
  int boff = hi ? 16 : 0;

  // Q A-fragments for hd chunks [0..31], [32..63]  (padded rows reuse last row)
  int l = qt * 16 + m;
  if (l >= Ll) l = Ll - 1;
  const __bf16* qrow = q + ((size_t)l * Nn + n) * Dm + h * HDm;
  v8bf q00 = *(const v8bf*)(qrow + aoff);
  v8bf q01 = *(const v8bf*)(qrow + 8 + aoff);
  v8bf q10 = *(const v8bf*)(qrow + 32 + aoff);
  v8bf q11 = *(const v8bf*)(qrow + 40 + aoff);
  v16bf qa0 = __builtin_shufflevector(q00, q01, 0,1,2,3,4,5,6,7,8,9,10,11,12,13,14,15);
  v16bf qa1 = __builtin_shufflevector(q10, q11, 0,1,2,3,4,5,6,7,8,9,10,11,12,13,14,15);
  const float scale = 0.125f;  // 1/sqrt(64)

  // S tiles over 19 key tiles of 16
  for (int j = 0; j < 19; j++) {
    int key = j * 16 + m;
    int kc  = key < Ll ? key : Ll - 1;
    const __bf16* krow = k + ((size_t)kc * Nn + n) * Dm + h * HDm + boff;
    v16bf kb0 = *(const v16bf*)(krow);
    v16bf kb1 = *(const v16bf*)(krow + 32);
    v8f s = {};
    s = __builtin_amdgcn_wmma_f32_16x16x32_bf16(false, qa0, false, kb0, (short)0, s, false, false);
    s = __builtin_amdgcn_wmma_f32_16x16x32_bf16(false, qa1, false, kb1, (short)0, s, false, false);
#pragma unroll
    for (int vg = 0; vg < 8; vg++) {
      int rm  = vg + hi * 8;
      int col = j * 16 + m;
      S[rm * 320 + col] = (col < Ll) ? s[vg] * scale : -1e30f;
    }
  }
  // pad columns 304..319 so the P*V loop can run 10 full chunks of 32
  for (int tix = lane; tix < 256; tix += 32)
    S[(tix >> 4) * 320 + 304 + (tix & 15)] = -1e30f;
  asm volatile("s_wait_dscnt 0" ::: "memory");

  // softmax: lane r < 16 owns row r
  if (lane < 16) {
    float mx = -1e30f;
    for (int c = 0; c < 320; c++) mx = fmaxf(mx, S[lane * 320 + c]);
    float sum = 0.f;
    for (int c = 0; c < 320; c++) {
      float e = expf(S[lane * 320 + c] - mx);
      S[lane * 320 + c] = e;
      sum += e;
    }
    float inv = 1.f / sum;
    for (int c = 0; c < 320; c++) S[lane * 320 + c] *= inv;
  }
  asm volatile("s_wait_dscnt 0" ::: "memory");

  // O = P V : 10 chunks of 32 keys, 4 hd column tiles of 16
  const __bf16 zb = __builtin_bit_cast(__bf16, (unsigned short)0);
  v8f o0 = {}, o1 = {}, o2 = {}, o3 = {};
  for (int jc = 0; jc < 10; jc++) {
    v16bf pa;
    int kb = jc * 32 + aoff;
#pragma unroll
    for (int i = 0; i < 8; i++) {
      pa[i]     = f2bf(S[m * 320 + kb + i]);
      pa[8 + i] = f2bf(S[m * 320 + kb + 8 + i]);
    }
    int vbk = jc * 32 + boff;
    v16bf b0, b1, b2, b3;
#pragma unroll
    for (int i = 0; i < 16; i++) {
      int key = vbk + i;
      bool in = key < Ll;
      size_t off = ((size_t)(in ? key : 0) * Nn + n) * Dm + h * HDm + m;
      b0[i] = in ? v[off]      : zb;
      b1[i] = in ? v[off + 16] : zb;
      b2[i] = in ? v[off + 32] : zb;
      b3[i] = in ? v[off + 48] : zb;
    }
    o0 = __builtin_amdgcn_wmma_f32_16x16x32_bf16(false, pa, false, b0, (short)0, o0, false, false);
    o1 = __builtin_amdgcn_wmma_f32_16x16x32_bf16(false, pa, false, b1, (short)0, o1, false, false);
    o2 = __builtin_amdgcn_wmma_f32_16x16x32_bf16(false, pa, false, b2, (short)0, o2, false, false);
    o3 = __builtin_amdgcn_wmma_f32_16x16x32_bf16(false, pa, false, b3, (short)0, o3, false, false);
  }

  int rb = qt * 16 + hi * 8;
#pragma unroll
  for (int vg = 0; vg < 8; vg++) {
    int r = rb + vg;
    if (r < Ll) {
      __bf16* orow = ao + ((size_t)r * Nn + n) * Dm + h * HDm + m;
      orow[0]  = f2bf(o0[vg]);
      orow[16] = f2bf(o1[vg]);
      orow[32] = f2bf(o2[vg]);
      orow[48] = f2bf(o3[vg]);
    }
  }
}

// ---------------------------------------------------------------------------
extern "C" void kernel_launch(void* const* d_in, const int* in_sizes, int n_in,
                              void* d_out, int out_size, void* d_ws, size_t ws_size,
                              hipStream_t stream)
{
  const float* x    = (const float*)d_in[0];
  const float* ln1w = (const float*)d_in[1];
  const float* ln1b = (const float*)d_in[2];
  const float* wq   = (const float*)d_in[3];
  const float* bq   = (const float*)d_in[4];
  const float* wk   = (const float*)d_in[5];
  const float* bk   = (const float*)d_in[6];
  const float* wv   = (const float*)d_in[7];
  const float* bv   = (const float*)d_in[8];
  const float* wo   = (const float*)d_in[9];
  const float* bo   = (const float*)d_in[10];
  const float* ln2w = (const float*)d_in[11];
  const float* ln2b = (const float*)d_in[12];
  const float* mw1  = (const float*)d_in[13];
  const float* mb1  = (const float*)d_in[14];
  const float* mw2  = (const float*)d_in[15];
  const float* mb2  = (const float*)d_in[16];
  const float* ew1  = (const float*)d_in[17];
  const float* eb1  = (const float*)d_in[18];
  const float* ew2  = (const float*)d_in[19];
  const float* eb2  = (const float*)d_in[20];
  float* out = (float*)d_out;

  // ---- workspace carving (256B-aligned) ----
  char* p = (char*)d_ws;
  auto carve = [&](size_t bytes) -> char* {
    char* r = p;
    p += (bytes + 255) & ~(size_t)255;
    return r;
  };
  const size_t TD = (size_t)Ttok * Dm;
  __bf16* t_bf  = (__bf16*)carve(TD * 2);
  __bf16* q_bf  = (__bf16*)carve(TD * 2);
  __bf16* k_bf  = (__bf16*)carve(TD * 2);
  __bf16* v_bf  = (__bf16*)carve(TD * 2);
  __bf16* ao_bf = (__bf16*)carve(TD * 2);
  float*  x2    = (float*)carve(TD * 4);
  __bf16* tem_bf= (__bf16*)carve(TD * 2);
  __bf16* h1_bf = (__bf16*)carve((size_t)(257 * Nn) * Mm * 2);
  __bf16* he_bf = (__bf16*)carve((size_t)Ee * Nn * Mm * 2);
  __bf16* wq_bf = (__bf16*)carve((size_t)Dm * Dm * 2);
  __bf16* wk_bf = (__bf16*)carve((size_t)Dm * Dm * 2);
  __bf16* wv_bf = (__bf16*)carve((size_t)Dm * Dm * 2);
  __bf16* wo_bf = (__bf16*)carve((size_t)Dm * Dm * 2);
  __bf16* mw1_bf= (__bf16*)carve((size_t)Mm * Dm * 2);
  __bf16* mw2_bf= (__bf16*)carve((size_t)Dm * Mm * 2);
  __bf16* ew1_bf= (__bf16*)carve((size_t)Ee * Mm * Dm * 2);
  __bf16* ew2_bf= (__bf16*)carve((size_t)Ee * Dm * Mm * 2);

  dim3 blk32(32, 1, 1);
  auto cvt = [&](const float* src, __bf16* dst, long long n) {
    long long blocks = (n + 8LL * 256 - 1) / (8LL * 256);
    cvtbf_kernel<<<dim3((unsigned)blocks), 256, 0, stream>>>(src, dst, n);
  };

  // ---- weight conversion passes (fp32 -> bf16, once per launch) ----
  cvt(wq, wq_bf, (long long)Dm * Dm);
  cvt(wk, wk_bf, (long long)Dm * Dm);
  cvt(wv, wv_bf, (long long)Dm * Dm);
  cvt(wo, wo_bf, (long long)Dm * Dm);
  cvt(mw1, mw1_bf, (long long)Mm * Dm);
  cvt(mw2, mw2_bf, (long long)Dm * Mm);
  cvt(ew1, ew1_bf, (long long)Ee * Mm * Dm);
  cvt(ew2, ew2_bf, (long long)Ee * Dm * Mm);

  // ---- LN1 -> bf16 ----
  ln_kernel<<<dim3((Ttok + 7) / 8), 256, 0, stream>>>(x, ln1w, ln1b, t_bf, Ttok);

  // ---- Q, K, V projections (bf16 out) ----
  gemm_bf16_kernel<<<dim3(Ttok / 16, Dm / 64, 1), blk32, 0, stream>>>(
      t_bf, wq_bf, bq, nullptr, nullptr, q_bf, Dm, Dm, 0, 0, 0, 0, 0, 0);
  gemm_bf16_kernel<<<dim3(Ttok / 16, Dm / 64, 1), blk32, 0, stream>>>(
      t_bf, wk_bf, bk, nullptr, nullptr, k_bf, Dm, Dm, 0, 0, 0, 0, 0, 0);
  gemm_bf16_kernel<<<dim3(Ttok / 16, Dm / 64, 1), blk32, 0, stream>>>(
      t_bf, wv_bf, bv, nullptr, nullptr, v_bf, Dm, Dm, 0, 0, 0, 0, 0, 0);

  // ---- Attention (bf16 in/out) ----
  attn_kernel<<<dim3(19, Hh, Nn), blk32, 0, stream>>>(q_bf, k_bf, v_bf, ao_bf);

  // ---- O projection + residual(x) -> x2 (fp32) ----
  gemm_bf16_kernel<<<dim3(Ttok / 16, Dm / 64, 1), blk32, 0, stream>>>(
      ao_bf, wo_bf, bo, x, x2, nullptr, Dm, Dm, 0, 0, 0, 0, 0, 0);

  // ---- LN2 -> bf16 ----
  ln_kernel<<<dim3((Ttok + 7) / 8), 256, 0, stream>>>(x2, ln2w, ln2b, tem_bf, Ttok);

  // ---- Dense MLP on first 257 sequence rows (16448 tokens) ----
  const int T1 = 257 * Nn;
  gemm_bf16_kernel<<<dim3(T1 / 16, Mm / 64, 1), blk32, 0, stream>>>(
      tem_bf, mw1_bf, mb1, nullptr, nullptr, h1_bf, Dm, Mm, 0, 0, 0, 0, 0, 1);
  gemm_bf16_kernel<<<dim3(T1 / 16, Dm / 64, 1), blk32, 0, stream>>>(
      h1_bf, mw2_bf, mb2, x2, out, nullptr, Mm, Dm, 0, 0, 0, 0, 0, 0);

  // ---- Experts: rows 257..292, expert z handles 64 tokens ----
  const long long ND = (long long)Nn * Dm;
  gemm_bf16_kernel<<<dim3(Nn / 16, Mm / 64, Ee), blk32, 0, stream>>>(
      tem_bf + (size_t)257 * ND, ew1_bf, eb1, nullptr, nullptr, he_bf, Dm, Mm,
      ND, (long long)Mm * Dm, (long long)Mm, 0, (long long)Nn * Mm, 1);
  gemm_bf16_kernel<<<dim3(Nn / 16, Dm / 64, Ee), blk32, 0, stream>>>(
      he_bf, ew2_bf, eb2, x2 + (size_t)257 * ND, out + (size_t)257 * ND, nullptr, Mm, Dm,
      (long long)Nn * Mm, (long long)Dm * Mm, (long long)Dm, ND, ND, 0);

  (void)in_sizes; (void)n_in; (void)out_size; (void)ws_size;
}
